// PointTransformerLayer_71820443124429
// MI455X (gfx1250) — matrix operand order repaired and checked by reading, not compiled, not run on hardware
//
#include <hip/hip_runtime.h>

// ---------------- problem constants (from reference setup_inputs) -----------
constexpr int B_ = 2;
constexpr int N_ = 4096;
constexpr int D_ = 256;
constexpr int H_ = 64;
constexpr int K_ = 16;
constexpr int Q_ = 4;     // queries per block in fused kernel

typedef __attribute__((ext_vector_type(16))) __bf16 v16bf;
typedef __attribute__((ext_vector_type(8)))  float  v8f;
typedef __attribute__((ext_vector_type(4)))  float  f32x4;

union Frag { v16bf v; f32x4 f[2]; };

// A fragment: row `lane%16`, K chunks {0..7,16..23} (lanes<16) / {8..15,24..31}
__device__ inline v16bf load_a(const __bf16* __restrict__ row, int kk, int hs) {
    Frag a;
    const int ka = kk * 32 + (hs ? 8 : 0);
    a.f[0] = *(const f32x4*)(row + ka);
    a.f[1] = *(const f32x4*)(row + ka + 16);
    return a.v;
}
// B fragment: col `lane%16`, K chunks {0..7,16..23} (lanes<16) / {8..15,24..31} (B(k,o)=W[o][k], row-major W)
__device__ inline v16bf load_b(const __bf16* __restrict__ row, int kk, int hs) {
    Frag b;
    const int kb = kk * 32 + (hs ? 8 : 0);
    b.f[0] = *(const f32x4*)(row + kb);
    b.f[1] = *(const f32x4*)(row + kb + 16);
    return b.v;
}

// ---------------------------------------------------------------------------
// Kernel 1: fp32 -> bf16 conversion (activations + weights)
// ---------------------------------------------------------------------------
__global__ __launch_bounds__(256) void cvt_f32_bf16(const float* __restrict__ src,
                                                    __bf16* __restrict__ dst, int n) {
    int i = blockIdx.x * 256 + threadIdx.x;
    if (i < n) dst[i] = (__bf16)src[i];
}

// ---------------------------------------------------------------------------
// Kernel 2: brute-force KNN, k=16.  Whole batch cloud staged in LDS (SoA +
// |p|^2 = 64 KB); register-resident top-16 via unrolled insert+bubble
// (strict '<' preserves top_k's earlier-index-first tie ordering).
// ---------------------------------------------------------------------------
__global__ __launch_bounds__(256) void knn_kernel(const float* __restrict__ xyz,
                                                  int* __restrict__ knn_idx) {
    __shared__ float sx[N_], sy[N_], sz[N_], sq[N_];
    const int b     = blockIdx.x / (N_ / 256);
    const int qbase = (blockIdx.x % (N_ / 256)) * 256;
    const float* base = xyz + (size_t)b * N_ * 3;
    for (int i = threadIdx.x; i < N_; i += 256) {
        float x = base[i * 3 + 0], y = base[i * 3 + 1], z = base[i * 3 + 2];
        sx[i] = x; sy[i] = y; sz[i] = z; sq[i] = x * x + y * y + z * z;
    }
    __syncthreads();
    const int qn = qbase + threadIdx.x;
    const float px = sx[qn], py = sy[qn], pz = sz[qn], ps = sq[qn];
    float bd[K_]; int bi[K_];
#pragma unroll
    for (int j = 0; j < K_; ++j) { bd[j] = 3.4e38f; bi[j] = 0; }
    for (int m = 0; m < N_; ++m) {
        float d = ps + sq[m] - 2.0f * (px * sx[m] + py * sy[m] + pz * sz[m]);
        if (d < bd[K_ - 1]) {
            bd[K_ - 1] = d; bi[K_ - 1] = m;
#pragma unroll
            for (int j = K_ - 1; j > 0; --j) {
                if (bd[j] < bd[j - 1]) {
                    float td = bd[j]; bd[j] = bd[j - 1]; bd[j - 1] = td;
                    int   ti = bi[j]; bi[j] = bi[j - 1]; bi[j - 1] = ti;
                }
            }
        }
    }
    int* o = knn_idx + (size_t)(b * N_ + qn) * K_;
#pragma unroll
    for (int j = 0; j < K_; ++j) o[j] = bi[j];
}

// ---------------------------------------------------------------------------
// Kernel 3: q / psi(x) / alpha(x) = x @ W^T + b, bf16 WMMA.
// One wave = 4 row-tiles (M=64) x 1 col-tile, with the whole 256x16 B strip
// register-resident (8 Frags = 64 VGPRs) -> 4x less weight traffic.
// jobs = 3 mats * 128 row-groups * 16 col-tiles = 6144 waves -> 768 blocks.
// ---------------------------------------------------------------------------
__global__ __launch_bounds__(256) void qkv_gemm(
    const __bf16* __restrict__ xb,
    const __bf16* __restrict__ Wphi, const __bf16* __restrict__ Wpsi,
    const __bf16* __restrict__ Walp,
    const float* __restrict__ bphi, const float* __restrict__ bpsi,
    const float* __restrict__ balp,
    float* __restrict__ q, float* __restrict__ psix, float* __restrict__ alphax) {
    const int wave = threadIdx.x >> 5;
    const int lane = threadIdx.x & 31;
    const int t    = blockIdx.x * 8 + wave;
    const int mat  = t / (128 * 16);
    const int rem  = t % (128 * 16);
    const int rg   = rem >> 4;          // 64-row group
    const int ct   = rem & 15;
    const __bf16* W    = (mat == 0) ? Wphi : (mat == 1) ? Wpsi : Walp;
    const float*  bias = (mat == 0) ? bphi : (mat == 1) ? bpsi : balp;
    float*        out  = (mat == 0) ? q    : (mat == 1) ? psix : alphax;

    const int hs = lane >> 4;
    const int lm = lane & 15;
    const __bf16* brow = W + (size_t)(ct * 16 + lm) * D_;

    v16bf barr[8];
#pragma unroll
    for (int kk = 0; kk < 8; ++kk) barr[kk] = load_b(brow, kk, hs);

    const int col = ct * 16 + lm;
    const float bv = bias[col];
#pragma unroll
    for (int rt = 0; rt < 4; ++rt) {
        const __bf16* arow = xb + (size_t)(rg * 64 + rt * 16 + lm) * D_;
        v8f acc = {};
#pragma unroll
        for (int kk = 0; kk < 8; ++kk) {
            v16bf a = load_a(arow, kk, hs);
            acc = __builtin_amdgcn_wmma_f32_16x16x32_bf16(false, a, false, barr[kk],
                                                          (short)0, acc, false, false);
        }
#pragma unroll
        for (int r = 0; r < 8; ++r) {
            int row = rg * 64 + rt * 16 + r + hs * 8;
            out[(size_t)row * D_ + col] = acc[r] + bv;
        }
    }
}

// ---------------------------------------------------------------------------
// Kernel 4: fused attention, Q_=4 queries per block (256 thr / 8 waves).
// Each wave owns col-tiles {w, w+8}; B strips stay in registers across the
// 4 query A-tiles.  Softmax over k fused into the g2 epilogue via
// __shfl_xor(.,16) half-wave reductions (C/D layout: lane L / L^16 hold the
// two 8-row halves of one column).
// ---------------------------------------------------------------------------
__global__ __launch_bounds__(256) void fused_attn(
    const float* __restrict__ xyz, const int* __restrict__ knn_idx,
    const float* __restrict__ q, const float* __restrict__ psix,
    const float* __restrict__ alphax,
    const __bf16* __restrict__ Wg1b, const __bf16* __restrict__ Wg2b,
    const __bf16* __restrict__ Wp2b,
    const float* __restrict__ Wp1, const float* __restrict__ bp1,
    const float* __restrict__ bp2,
    const float* __restrict__ bg1, const float* __restrict__ bg2,
    float* __restrict__ out_feat, float* __restrict__ out_attn) {
    __shared__ int    s_idx[Q_][K_];
    __shared__ float  s_rel[Q_][K_][3];
    __shared__ __bf16 s_hid[Q_][K_][H_];      //  8 KB
    __shared__ __bf16 s_pe [Q_][K_][D_];      // 32 KB (bf16 pos_enc)
    __shared__ __bf16 s_g  [Q_][K_][D_];      // 32 KB
    __shared__ __bf16 s_h  [Q_][K_][D_];      // 32 KB

    const int bn0 = blockIdx.x * Q_;
    const int tid = threadIdx.x;

    // neighbors + rel for Q_*K_ = 64 slots
    if (tid < Q_ * K_) {
        const int qq = tid >> 4, j = tid & 15;
        const int bq = bn0 + qq;
        const int b  = bq / N_;
        int nb = knn_idx[(size_t)bq * K_ + j];
        s_idx[qq][j] = nb;
        const float* pq = xyz + (size_t)bq * 3;
        const float* pn = xyz + (size_t)(b * N_ + nb) * 3;
        s_rel[qq][j][0] = pq[0] - pn[0];
        s_rel[qq][j][1] = pq[1] - pn[1];
        s_rel[qq][j][2] = pq[2] - pn[2];
    }
    __syncthreads();

    // hidden = relu(rel @ Wp1^T + bp1): Q_*16*64 = 4096 outputs, 16/thread
    {
#pragma unroll
        for (int u = 0; u < 16; ++u) {
            int idx = tid * 16 + u;
            int qq = idx >> 10, j = (idx >> 6) & 15, h = idx & 63;
            float s = bp1[h] + Wp1[h * 3 + 0] * s_rel[qq][j][0]
                             + Wp1[h * 3 + 1] * s_rel[qq][j][1]
                             + Wp1[h * 3 + 2] * s_rel[qq][j][2];
            s_hid[qq][j][h] = (__bf16)(s > 0.f ? s : 0.f);
        }
    }
    __syncthreads();

    const int wave = tid >> 5, lane = tid & 31;
    const int hs = lane >> 4, lm = lane & 15;

    // pos_enc = hid @ Wp2^T + bp2 -> s_pe (bf16); B strip (K=64) resident
#pragma unroll
    for (int tt = 0; tt < 2; ++tt) {
        const int ct  = wave + tt * 8;
        const int col = ct * 16 + lm;
        const __bf16* brow = Wp2b + (size_t)col * H_;
        v16bf b0 = load_b(brow, 0, hs);
        v16bf b1 = load_b(brow, 1, hs);
        const float bv = bp2[col];
#pragma unroll
        for (int qq = 0; qq < Q_; ++qq) {
            const __bf16* arow = &s_hid[qq][lm][0];
            v8f acc = {};
            acc = __builtin_amdgcn_wmma_f32_16x16x32_bf16(false, load_a(arow, 0, hs),
                                                          false, b0, (short)0, acc, false, false);
            acc = __builtin_amdgcn_wmma_f32_16x16x32_bf16(false, load_a(arow, 1, hs),
                                                          false, b1, (short)0, acc, false, false);
#pragma unroll
            for (int r = 0; r < 8; ++r)
                s_pe[qq][r + hs * 8][col] = (__bf16)(acc[r] + bv);
        }
    }
    __syncthreads();

    // g = q[bq] - psix[nb_j] + pos_enc   (bf16 A operand)
    {
        const int j  = (tid >> 4) & 15;
        const int c0 = (tid & 15) * 16;
#pragma unroll
        for (int qq = 0; qq < Q_; ++qq) {
            const int bq = bn0 + qq;
            const int b  = bq / N_;
            const float* qr = q    + (size_t)bq * D_;
            const float* pr = psix + (size_t)(b * N_ + s_idx[qq][j]) * D_;
#pragma unroll
            for (int u = 0; u < 16; ++u) {
                int c = c0 + u;
                s_g[qq][j][c] = (__bf16)(qr[c] - pr[c] + (float)s_pe[qq][j][c]);
            }
        }
    }
    __syncthreads();

    // h = relu(g @ Wg1^T + bg1); B strip (K=256, 64 VGPRs) resident per tile
#pragma unroll
    for (int tt = 0; tt < 2; ++tt) {
        const int ct  = wave + tt * 8;
        const int col = ct * 16 + lm;
        const __bf16* brow = Wg1b + (size_t)col * D_;
        v16bf barr[8];
#pragma unroll
        for (int kk = 0; kk < 8; ++kk) barr[kk] = load_b(brow, kk, hs);
        const float bv = bg1[col];
#pragma unroll
        for (int qq = 0; qq < Q_; ++qq) {
            const __bf16* arow = &s_g[qq][lm][0];
            v8f acc = {};
#pragma unroll
            for (int kk = 0; kk < 8; ++kk)
                acc = __builtin_amdgcn_wmma_f32_16x16x32_bf16(false, load_a(arow, kk, hs),
                                                              false, barr[kk], (short)0,
                                                              acc, false, false);
#pragma unroll
            for (int r = 0; r < 8; ++r) {
                float v = acc[r] + bv;
                s_h[qq][r + hs * 8][col] = (__bf16)(v > 0.f ? v : 0.f);
            }
        }
    }
    __syncthreads();

    // attn = h @ Wg2^T + bg2, then fused per-channel softmax over k and the
    // weighted sum, all in registers + one half-wave shuffle pair.
#pragma unroll
    for (int tt = 0; tt < 2; ++tt) {
        const int ct  = wave + tt * 8;
        const int col = ct * 16 + lm;
        const __bf16* brow = Wg2b + (size_t)col * D_;
        v16bf barr[8];
#pragma unroll
        for (int kk = 0; kk < 8; ++kk) barr[kk] = load_b(brow, kk, hs);
        const float bv = bg2[col];
#pragma unroll
        for (int qq = 0; qq < Q_; ++qq) {
            const __bf16* arow = &s_h[qq][lm][0];
            v8f acc = {};
#pragma unroll
            for (int kk = 0; kk < 8; ++kk)
                acc = __builtin_amdgcn_wmma_f32_16x16x32_bf16(false, load_a(arow, kk, hs),
                                                              false, barr[kk], (short)0,
                                                              acc, false, false);
            // this lane holds rows j = hs*8 + r of column `col`
            float av[8];
            float mx = -3.4e38f;
#pragma unroll
            for (int r = 0; r < 8; ++r) {
                av[r] = (acc[r] + bv) * 0.0625f;           // / sqrt(256)
                mx = av[r] > mx ? av[r] : mx;
            }
            mx = fmaxf(mx, __shfl_xor(mx, 16, 32));
            float sum = 0.f;
#pragma unroll
            for (int r = 0; r < 8; ++r) { av[r] = __expf(av[r] - mx); sum += av[r]; }
            sum += __shfl_xor(sum, 16, 32);
            const float inv = 1.0f / sum;

            const int bq = bn0 + qq;
            const int b  = bq / N_;
            float* oa = out_attn + ((size_t)bq * K_ + hs * 8) * D_ + col;
            float accum = 0.f;
#pragma unroll
            for (int r = 0; r < 8; ++r) {
                float w = av[r] * inv;
                oa[(size_t)r * D_] = w;
                int nb = s_idx[qq][hs * 8 + r];
                accum += w * (alphax[(size_t)(b * N_ + nb) * D_ + col]
                              + (float)s_pe[qq][hs * 8 + r][col]);
            }
            accum += __shfl_xor(accum, 16, 32);
            if (lane < 16) out_feat[(size_t)bq * D_ + col] = accum;
        }
    }
}

// ---------------------------------------------------------------------------
extern "C" void kernel_launch(void* const* d_in, const int* in_sizes, int n_in,
                              void* d_out, int out_size, void* d_ws, size_t ws_size,
                              hipStream_t stream) {
    const float* input_feature = (const float*)d_in[0];
    const float* xyz    = (const float*)d_in[1];
    const float* W_phi  = (const float*)d_in[2];
    const float* b_phi  = (const float*)d_in[3];
    const float* W_psi  = (const float*)d_in[4];
    const float* b_psi  = (const float*)d_in[5];
    const float* W_alp  = (const float*)d_in[6];
    const float* b_alp  = (const float*)d_in[7];
    const float* W_g1   = (const float*)d_in[8];
    const float* b_g1   = (const float*)d_in[9];
    const float* W_g2   = (const float*)d_in[10];
    const float* b_g2   = (const float*)d_in[11];
    const float* W_p1   = (const float*)d_in[12];
    const float* b_p1   = (const float*)d_in[13];
    const float* W_p2   = (const float*)d_in[14];
    const float* b_p2   = (const float*)d_in[15];
    (void)in_sizes; (void)n_in; (void)out_size; (void)ws_size;

    float* out_feat = (float*)d_out;                            // [B,N,D]
    float* out_attn = out_feat + (size_t)B_ * N_ * D_;          // [B,N,K,D]

    // workspace carve-up (~30.4 MB)
    char* ws = (char*)d_ws;
    int*    knn   = (int*)ws;            ws += (size_t)B_ * N_ * K_ * sizeof(int);
    __bf16* xb    = (__bf16*)ws;         ws += (size_t)B_ * N_ * D_ * sizeof(__bf16);
    __bf16* Wphib = (__bf16*)ws;         ws += (size_t)D_ * D_ * sizeof(__bf16);
    __bf16* Wpsib = (__bf16*)ws;         ws += (size_t)D_ * D_ * sizeof(__bf16);
    __bf16* Walpb = (__bf16*)ws;         ws += (size_t)D_ * D_ * sizeof(__bf16);
    __bf16* Wg1b  = (__bf16*)ws;         ws += (size_t)D_ * D_ * sizeof(__bf16);
    __bf16* Wg2b  = (__bf16*)ws;         ws += (size_t)D_ * D_ * sizeof(__bf16);
    __bf16* Wp2b  = (__bf16*)ws;         ws += (size_t)D_ * H_ * sizeof(__bf16);
    float*  qd    = (float*)ws;          ws += (size_t)B_ * N_ * D_ * sizeof(float);
    float*  psix  = (float*)ws;          ws += (size_t)B_ * N_ * D_ * sizeof(float);
    float*  alphx = (float*)ws;          ws += (size_t)B_ * N_ * D_ * sizeof(float);

    const int BND = B_ * N_ * D_;
    cvt_f32_bf16<<<(BND + 255) / 256, 256, 0, stream>>>(input_feature, xb, BND);
    cvt_f32_bf16<<<(D_ * D_ + 255) / 256, 256, 0, stream>>>(W_phi, Wphib, D_ * D_);
    cvt_f32_bf16<<<(D_ * D_ + 255) / 256, 256, 0, stream>>>(W_psi, Wpsib, D_ * D_);
    cvt_f32_bf16<<<(D_ * D_ + 255) / 256, 256, 0, stream>>>(W_alp, Walpb, D_ * D_);
    cvt_f32_bf16<<<(D_ * D_ + 255) / 256, 256, 0, stream>>>(W_g1, Wg1b, D_ * D_);
    cvt_f32_bf16<<<(D_ * D_ + 255) / 256, 256, 0, stream>>>(W_g2, Wg2b, D_ * D_);
    cvt_f32_bf16<<<(D_ * H_ + 255) / 256, 256, 0, stream>>>(W_p2, Wp2b, D_ * H_);

    knn_kernel<<<B_ * (N_ / 256), 256, 0, stream>>>(xyz, knn);

    // 3 mats * 128 row-groups * 16 col-tiles = 6144 waves -> 768 blocks
    qkv_gemm<<<768, 256, 0, stream>>>(xb, Wphib, Wpsib, Walpb,
                                      b_phi, b_psi, b_alp, qd, psix, alphx);

    fused_attn<<<(B_ * N_) / Q_, 256, 0, stream>>>(xyz, knn, qd, psix, alphx,
                                                   Wg1b, Wg2b, Wp2b,
                                                   W_p1, b_p1, b_p2, b_g1, b_g2,
                                                   out_feat, out_attn);
}
